// KPRN_59304908423380
// MI455X (gfx1250) — matrix operand order repaired and checked by reading, not compiled, not run on hardware
//
#include <hip/hip_runtime.h>
#include <hip/hip_bf16.h>
#include <cstdint>

// ---------------------------------------------------------------------------
// Problem constants (match reference)
// ---------------------------------------------------------------------------
#define BB     1024
#define PP     20
#define LL     3
#define TT     5
#define EE     128
#define DDOC   768
#define DENT   100
#define NSEQ   (BB * PP)          // 20480
#define NROWS3 (NSEQ * LL)        // 61440 entity / relation rows
#define GAMMA_INV 0.5f
#define EPSC   1e-7f

// LDS strides in halves (padded: 16B-aligned rows, +4 dword skew for banks)
#define SX 264   // X staging: 256 + 8
#define SH 136   // h / 128-col staging: 128 + 8
#define SD 776   // doc staging: 768 + 8

typedef __attribute__((ext_vector_type(16))) _Float16 v16h;
typedef __attribute__((ext_vector_type(8)))  _Float16 v8h;
typedef __attribute__((ext_vector_type(8)))  float    v8f;

union V8F { v8f v; float f[8]; };

__device__ __forceinline__ v16h cat8(v8h a, v8h b) {
  return __builtin_shufflevector(a, b, 0,1,2,3,4,5,6,7,8,9,10,11,12,13,14,15);
}

// A fragment: 16x32 f16, row-major source, per ISA 7.12.2 layout.
// lane<16 -> K {0..7,16..23}, lane>=16 -> K {8..15,24..31}
__device__ __forceinline__ v16h frag_a(const _Float16* base, int strideH, int kb, int lane) {
  const int m    = lane & 15;
  const int koff = (lane < 16) ? 0 : 8;
  const _Float16* p = base + m * strideH + kb + koff;
  return cat8(*(const v8h*)p, *(const v8h*)(p + 16));
}

// B fragment: 32x16 f16 from weight stored [n_out][k] (K contiguous per row).
// lane<16 -> K {0..7,16..23} of column lane, lane>=16 -> K {8..15,24..31} of column lane-16.
__device__ __forceinline__ v16h frag_b(const _Float16* base, int strideH, int kb, int lane) {
  const int n    = lane & 15;
  const int koff = (lane < 16) ? 0 : 8;
  const _Float16* p = base + n * strideH + kb + koff;
  return cat8(*(const v8h*)p, *(const v8h*)(p + 16));
}

__device__ __forceinline__ v8f wmma16(v16h a, v16h b, v8f c) {
  return __builtin_amdgcn_wmma_f32_16x16x32_f16(false, a, false, b, (short)0, c, false, false);
}

__device__ __forceinline__ float sigf(float x) { return 1.0f / (1.0f + __expf(-x)); }

// CDNA5 async global->LDS copy (16B per lane), tracked by ASYNCcnt.
// LDS byte address = low 32 bits of the flat shared pointer (shared aperture
// occupies the high dword; addr[31:0] is the LDS address per ISA 10.2).
__device__ __forceinline__ uint32_t lds_addr32(const void* p) {
  return (uint32_t)(uintptr_t)p;
}
__device__ __forceinline__ void async_g2l_b128(uint32_t lds_off, const void* gaddr) {
  asm volatile("global_load_async_to_lds_b128 %0, %1, off"
               :: "v"(lds_off), "v"(gaddr) : "memory");
}
__device__ __forceinline__ void wait_asynccnt0() {
  asm volatile("s_wait_asynccnt 0" ::: "memory");
}

// ---------------------------------------------------------------------------
// Kernel 1: weight conversion / transposition to f16 [n_out][k] layout
// ---------------------------------------------------------------------------
#define SEG0 (128 * 768)   // nc_W1_T
#define SEG1 (128 * 128)   // nc_W2_T
#define SEG2 (128 * 128)   // ec_W_T (K padded to 128)
#define SEG3 (128 * 128)   // rc_W_T
#define SEG4 (512 * 256)   // W_ih copy
#define SEG5 (512 * 128)   // W_hh copy
#define SEG6 (128 * 128)   // mlp_W1_T
#define SEG7 (512)         // bias_sum
#define PREP_TOTAL (SEG0+SEG1+SEG2+SEG3+SEG4+SEG5+SEG6+SEG7)

__global__ __launch_bounds__(256) void prep_kernel(
    const float* nc_W1, const float* nc_W2, const float* ec_W, const float* rc_W,
    const float* W_ih, const float* W_hh, const float* mlp_W1,
    const float* b_ih, const float* b_hh,
    _Float16* ncW1T, _Float16* ncW2T, _Float16* ecWT, _Float16* rcWT,
    _Float16* WihH, _Float16* WhhH, _Float16* mlpW1T, float* bsum)
{
  int i = blockIdx.x * blockDim.x + threadIdx.x;
  if (i >= PREP_TOTAL) return;
  if (i < SEG0) {                       // nc_W1 (768,128) -> [n][k] f16
    int n = i / 768, k = i % 768;
    ncW1T[i] = (_Float16)nc_W1[k * 128 + n];
    return;
  }
  i -= SEG0;
  if (i < SEG1) {
    int n = i / 128, k = i % 128;
    ncW2T[i] = (_Float16)nc_W2[k * 128 + n];
    return;
  }
  i -= SEG1;
  if (i < SEG2) {
    int n = i / 128, k = i % 128;
    ecWT[i] = (k < DENT) ? (_Float16)ec_W[k * 128 + n] : (_Float16)0.0f;
    return;
  }
  i -= SEG2;
  if (i < SEG3) {
    int n = i / 128, k = i % 128;
    rcWT[i] = (k < DENT) ? (_Float16)rc_W[k * 128 + n] : (_Float16)0.0f;
    return;
  }
  i -= SEG3;
  if (i < SEG4) { WihH[i] = (_Float16)W_ih[i]; return; }   // already [n][k]
  i -= SEG4;
  if (i < SEG5) { WhhH[i] = (_Float16)W_hh[i]; return; }   // already [n][k]
  i -= SEG5;
  if (i < SEG6) {
    int n = i / 128, k = i % 128;
    mlpW1T[i] = (_Float16)mlp_W1[k * 128 + n];
    return;
  }
  i -= SEG6;
  if (i < SEG7) { bsum[i] = b_ih[i] + b_hh[i]; }
}

// ---------------------------------------------------------------------------
// Kernel 2: news_compress  (2048 rows = [item1 ; item2])
//   tanh( elu( doc @ W1 + b1 ) @ W2 + b2 )  -> fp32 [2048][128]
// ---------------------------------------------------------------------------
__global__ __launch_bounds__(256) void news_kernel(
    const int* item1, const int* item2, const float* doc_table,
    const _Float16* W1T, const float* b1,
    const _Float16* W2T, const float* b2,
    float* news_emb)
{
  __shared__ __align__(16) _Float16 Ds[16 * SD];
  __shared__ __align__(16) _Float16 H1[16 * SH];
  const int tid = threadIdx.x;
  const int n0  = blockIdx.x * 16;

  for (int i = tid; i < 16 * 768; i += 256) {
    int r = i / 768, k = i % 768;
    int rr = n0 + r;
    int doc = (rr < BB) ? item1[rr] : item2[rr - BB];
    Ds[r * SD + k] = (_Float16)doc_table[(long)doc * DDOC + k];
  }
  __syncthreads();

  const int lane = tid & 31;
  const int j0   = (tid >> 5) * 16;
  const int colb = j0 + (lane & 15);

  V8F acc; float bv = b1[colb];
#pragma unroll
  for (int c = 0; c < 8; c++) acc.f[c] = bv;
#pragma unroll
  for (int kb = 0; kb < 768; kb += 32) {
    v16h a = frag_a(Ds, SD, kb, lane);
    v16h b = frag_b(W1T + j0 * 768, 768, kb, lane);
    acc.v = wmma16(a, b, acc.v);
  }
#pragma unroll
  for (int c = 0; c < 8; c++) {              // ELU
    float x = acc.f[c];
    float e = (x > 0.0f) ? x : (__expf(x) - 1.0f);
    int m = c + ((lane < 16) ? 0 : 8);
    H1[m * SH + colb] = (_Float16)e;
  }
  __syncthreads();

  V8F o; bv = b2[colb];
#pragma unroll
  for (int c = 0; c < 8; c++) o.f[c] = bv;
#pragma unroll
  for (int kb = 0; kb < 128; kb += 32) {
    v16h a = frag_a(H1, SH, kb, lane);
    v16h b = frag_b(W2T + j0 * 128, 128, kb, lane);
    o.v = wmma16(a, b, o.v);
  }
#pragma unroll
  for (int c = 0; c < 8; c++) {
    int m = c + ((lane < 16) ? 0 : 8);
    news_emb[(long)(n0 + m) * 128 + colb] = tanhf(o.f[c]);
  }
}

// ---------------------------------------------------------------------------
// Kernel 3: entity / relation compress: tanh( table[idx] @ W + b ) -> f16
//   idx == nullptr -> id 0 (for the padded-edge relation embedding)
// ---------------------------------------------------------------------------
__global__ __launch_bounds__(256) void embed_kernel(
    const float* table, const int* idx, int n_rows,
    const _Float16* WT, const float* bias, _Float16* out)
{
  __shared__ __align__(16) _Float16 As[16 * SH];
  const int tid = threadIdx.x;
  const int n0  = blockIdx.x * 16;

  for (int i = tid; i < 16 * 128; i += 256) {
    int r = i / 128, k = i % 128;
    int row = n0 + r;
    float v = 0.0f;
    if (row < n_rows && k < DENT) {
      int id = idx ? idx[row] : 0;
      v = table[(long)id * DENT + k];
    }
    As[r * SH + k] = (_Float16)v;
  }
  __syncthreads();

  const int lane = tid & 31;
  const int j0   = (tid >> 5) * 16;
  const int colb = j0 + (lane & 15);

  V8F acc; float bv = bias[colb];
#pragma unroll
  for (int c = 0; c < 8; c++) acc.f[c] = bv;
#pragma unroll
  for (int kb = 0; kb < 128; kb += 32) {
    v16h a = frag_a(As, SH, kb, lane);
    v16h b = frag_b(WT + j0 * 128, 128, kb, lane);
    acc.v = wmma16(a, b, acc.v);
  }
#pragma unroll
  for (int c = 0; c < 8; c++) {
    int m = c + ((lane < 16) ? 0 : 8);
    int row = n0 + m;
    if (row < n_rows)
      out[(long)row * 128 + colb] = (_Float16)tanhf(acc.f[c]);
  }
}

// ---------------------------------------------------------------------------
// Kernel 4: fused LSTM (T=5) + MLP head. One WG = 16 sequences, 8 waves.
// Wave w owns hidden columns [16w,16w+16); computes i/f/g/o tiles via WMMA,
// keeps c-state in v8f registers, shares h via LDS f16 between steps.
// f16 inputs are staged with GLOBAL_LOAD_ASYNC_TO_LDS_B128 (ASYNCcnt).
// ---------------------------------------------------------------------------
__global__ __launch_bounds__(256) void lstm_kernel(
    const float* news_emb, const _Float16* node_h, const _Float16* rel_h,
    const _Float16* rel0,
    const _Float16* WihH, const _Float16* WhhH, const float* bsum,
    const _Float16* mlpW1T, const float* mlp_b1,
    const float* mlp_W2, const float* mlp_b2,
    float* scores)
{
  __shared__ __align__(16) _Float16 Xs[TT * 16 * SX];
  __shared__ __align__(16) _Float16 Hs[16 * SH];
  const int tid = threadIdx.x;
  const int n0  = blockIdx.x * 16;

  // ---- async f16 staging: 8 segments x 16 rows x 16 chunks of 16B ----
  //  seg 0..2 : node part t=1..3   (node_h)
  //  seg 3..5 : rel  part t=0..2   (rel_h)
  //  seg 6..7 : rel  part t=3..4   (rel0, broadcast row)
  for (int i = tid; i < 8 * 16 * 16; i += 256) {
    int c = i & 15;            // 16B chunk within 128-half row part
    int r = (i >> 4) & 15;     // sequence row in tile
    int s = i >> 8;            // segment
    int n = n0 + r;
    const _Float16* src;
    _Float16* dst;
    if (s < 3) {               // node embeddings, t = s+1
      src = node_h + ((long)n * LL + s) * 128 + c * 8;
      dst = &Xs[((s + 1) * 16 + r) * SX + c * 8];
    } else if (s < 6) {        // relation embeddings, t = s-3
      int t = s - 3;
      src = rel_h + ((long)n * LL + t) * 128 + c * 8;
      dst = &Xs[(t * 16 + r) * SX + 128 + c * 8];
    } else {                   // rel-id-0 embedding, t = s-3 (3,4)
      int t = s - 3;
      src = rel0 + c * 8;
      dst = &Xs[(t * 16 + r) * SX + 128 + c * 8];
    }
    async_g2l_b128(lds_addr32(dst), src);
  }

  // ---- news rows (fp32 -> f16 conversion, regular path) ----
  for (int i = tid; i < 2 * 16 * 128; i += 256) {
    int k = i & 127;
    int r = (i >> 7) & 15;
    int hi = i >> 11;          // 0 -> t=0 (news1), 1 -> t=4 (news2)
    int n = n0 + r;
    int b = n / PP;
    float v = hi ? news_emb[(long)(BB + b) * 128 + k]
                 : news_emb[(long)b * 128 + k];
    Xs[((hi ? 4 : 0) * 16 + r) * SX + k] = (_Float16)v;
  }
  for (int i = tid; i < 16 * SH; i += 256) Hs[i] = (_Float16)0.0f;

  wait_asynccnt0();            // drain async LDS writes
  __syncthreads();

  const int lane = tid & 31;
  const int j0   = (tid >> 5) * 16;
  const int colb = j0 + (lane & 15);

  const float bi = bsum[0 * 128 + colb];
  const float bf = bsum[1 * 128 + colb];
  const float bg = bsum[2 * 128 + colb];
  const float bo = bsum[3 * 128 + colb];

  V8F cst;
#pragma unroll
  for (int c = 0; c < 8; c++) cst.f[c] = 0.0f;

  for (int t = 0; t < TT; t++) {
    V8F gi, gf, gg, go;
#pragma unroll
    for (int c = 0; c < 8; c++) { gi.f[c]=bi; gf.f[c]=bf; gg.f[c]=bg; go.f[c]=bo; }

    const _Float16* Xt = Xs + t * 16 * SX;
#pragma unroll
    for (int kb = 0; kb < 256; kb += 32) {     // x @ W_ih^T : A reused 4x
      v16h a = frag_a(Xt, SX, kb, lane);
      gi.v = wmma16(a, frag_b(WihH + (0*128 + j0) * 256, 256, kb, lane), gi.v);
      gf.v = wmma16(a, frag_b(WihH + (1*128 + j0) * 256, 256, kb, lane), gf.v);
      gg.v = wmma16(a, frag_b(WihH + (2*128 + j0) * 256, 256, kb, lane), gg.v);
      go.v = wmma16(a, frag_b(WihH + (3*128 + j0) * 256, 256, kb, lane), go.v);
    }
#pragma unroll
    for (int kb = 0; kb < 128; kb += 32) {     // h @ W_hh^T
      v16h a = frag_a(Hs, SH, kb, lane);
      gi.v = wmma16(a, frag_b(WhhH + (0*128 + j0) * 128, 128, kb, lane), gi.v);
      gf.v = wmma16(a, frag_b(WhhH + (1*128 + j0) * 128, 128, kb, lane), gf.v);
      gg.v = wmma16(a, frag_b(WhhH + (2*128 + j0) * 128, 128, kb, lane), gg.v);
      go.v = wmma16(a, frag_b(WhhH + (3*128 + j0) * 128, 128, kb, lane), go.v);
    }

    V8F hv;
#pragma unroll
    for (int c = 0; c < 8; c++) {              // gate nonlinearity + state update
      float iv = sigf(gi.f[c]);
      float fv = sigf(gf.f[c]);
      float gv = tanhf(gg.f[c]);
      float ov = sigf(go.f[c]);
      float cn = fv * cst.f[c] + iv * gv;
      cst.f[c] = cn;
      hv.f[c]  = ov * tanhf(cn);
    }
    __syncthreads();                           // all waves done reading h_prev
#pragma unroll
    for (int c = 0; c < 8; c++) {
      int m = c + ((lane < 16) ? 0 : 8);
      Hs[m * SH + colb] = (_Float16)hv.f[c];
    }
    __syncthreads();                           // h_new visible
  }

  // MLP layer 1: z = relu(h @ mlp_W1 + b1)
  V8F z; float bz = mlp_b1[colb];
#pragma unroll
  for (int c = 0; c < 8; c++) z.f[c] = bz;
#pragma unroll
  for (int kb = 0; kb < 128; kb += 32) {
    v16h a = frag_a(Hs, SH, kb, lane);
    z.v = wmma16(a, frag_b(mlpW1T + j0 * 128, 128, kb, lane), z.v);
  }
  _Float16* Zs = Xs;                           // reuse X staging as [16][SH]
#pragma unroll
  for (int c = 0; c < 8; c++) {
    int m = c + ((lane < 16) ? 0 : 8);
    Zs[m * SH + colb] = (_Float16)fmaxf(z.f[c], 0.0f);
  }
  __syncthreads();

  // MLP layer 2: scores = z @ mlp_W2 + b2  (128 -> 1)
  if (tid < 16) {
    float acc = mlp_b2[0];
    for (int k = 0; k < 128; k++)
      acc += (float)Zs[tid * SH + k] * mlp_W2[k];
    scores[n0 + tid] = acc;
  }
}

// ---------------------------------------------------------------------------
// Kernel 5: logsumexp over P, sigmoid, BCE loss (single 1024-thread block)
// ---------------------------------------------------------------------------
__global__ __launch_bounds__(1024) void loss_kernel(
    const float* label, const float* scores, float* out)
{
  __shared__ float red[1024];
  const int b = threadIdx.x;

  float mx = -3.0e38f;
  for (int p = 0; p < PP; p++)
    mx = fmaxf(mx, scores[b * PP + p] * GAMMA_INV);
  float se = 0.0f;
  for (int p = 0; p < PP; p++)
    se += __expf(scores[b * PP + p] * GAMMA_INV - mx);
  float lse  = mx + __logf(se);
  float pred = sigf(lse);
  out[1 + b] = pred;

  float pc = fminf(fmaxf(pred, EPSC), 1.0f - EPSC);
  float lb = label[b];
  red[b] = -(lb * __logf(pc) + (1.0f - lb) * __logf(1.0f - pc));
  __syncthreads();
  for (int s = 512; s > 0; s >>= 1) {
    if (b < s) red[b] += red[b + s];
    __syncthreads();
  }
  if (b == 0) out[0] = red[0] / (float)BB;
}

// ---------------------------------------------------------------------------
// Host launch
// ---------------------------------------------------------------------------
extern "C" void kernel_launch(void* const* d_in, const int* in_sizes, int n_in,
                              void* d_out, int out_size, void* d_ws, size_t ws_size,
                              hipStream_t stream) {
  const int*   item1     = (const int*)d_in[0];
  const int*   item2     = (const int*)d_in[1];
  const int*   paths     = (const int*)d_in[2];
  const int*   edges     = (const int*)d_in[3];
  const float* label     = (const float*)d_in[4];
  const float* doc_table = (const float*)d_in[5];
  const float* ent_table = (const float*)d_in[6];
  const float* rel_table = (const float*)d_in[7];
  const float* nc_W1 = (const float*)d_in[8];
  const float* nc_b1 = (const float*)d_in[9];
  const float* nc_W2 = (const float*)d_in[10];
  const float* nc_b2 = (const float*)d_in[11];
  const float* ec_W  = (const float*)d_in[12];
  const float* ec_b  = (const float*)d_in[13];
  const float* rc_W  = (const float*)d_in[14];
  const float* rc_b  = (const float*)d_in[15];
  const float* W_ih  = (const float*)d_in[16];
  const float* W_hh  = (const float*)d_in[17];
  const float* b_ih  = (const float*)d_in[18];
  const float* b_hh  = (const float*)d_in[19];
  const float* mlp_W1 = (const float*)d_in[20];
  const float* mlp_b1 = (const float*)d_in[21];
  const float* mlp_W2 = (const float*)d_in[22];
  const float* mlp_b2 = (const float*)d_in[23];

  char* ws = (char*)d_ws;
  auto alloc = [&](size_t bytes) -> char* {
    char* p = ws;
    ws += (bytes + 255) & ~(size_t)255;
    return p;
  };
  float*     news_emb = (float*)    alloc((size_t)2 * BB * 128 * 4);
  _Float16*  node_h   = (_Float16*) alloc((size_t)NROWS3 * 128 * 2);
  _Float16*  relE_h   = (_Float16*) alloc((size_t)NROWS3 * 128 * 2);
  _Float16*  rel0_h   = (_Float16*) alloc((size_t)16 * 128 * 2);
  _Float16*  ncW1T    = (_Float16*) alloc((size_t)128 * 768 * 2);
  _Float16*  ncW2T    = (_Float16*) alloc((size_t)128 * 128 * 2);
  _Float16*  ecWT     = (_Float16*) alloc((size_t)128 * 128 * 2);
  _Float16*  rcWT     = (_Float16*) alloc((size_t)128 * 128 * 2);
  _Float16*  WihH     = (_Float16*) alloc((size_t)512 * 256 * 2);
  _Float16*  WhhH     = (_Float16*) alloc((size_t)512 * 128 * 2);
  _Float16*  mlpW1T   = (_Float16*) alloc((size_t)128 * 128 * 2);
  float*     bsum     = (float*)    alloc((size_t)512 * 4);

  float* out     = (float*)d_out;
  float* scores  = out + 1 + BB;   // layout: [loss | predicts(B) | scores(B*P)]

  prep_kernel<<<(PREP_TOTAL + 255) / 256, 256, 0, stream>>>(
      nc_W1, nc_W2, ec_W, rc_W, W_ih, W_hh, mlp_W1, b_ih, b_hh,
      ncW1T, ncW2T, ecWT, rcWT, WihH, WhhH, mlpW1T, bsum);

  news_kernel<<<(2 * BB) / 16, 256, 0, stream>>>(
      item1, item2, doc_table, ncW1T, nc_b1, ncW2T, nc_b2, news_emb);

  embed_kernel<<<NROWS3 / 16, 256, 0, stream>>>(
      ent_table, paths, NROWS3, ecWT, ec_b, node_h);

  embed_kernel<<<NROWS3 / 16, 256, 0, stream>>>(
      rel_table, edges, NROWS3, rcWT, rc_b, relE_h);

  embed_kernel<<<1, 256, 0, stream>>>(
      rel_table, nullptr, 16, rcWT, rc_b, rel0_h);

  lstm_kernel<<<NSEQ / 16, 256, 0, stream>>>(
      news_emb, node_h, relE_h, rel0_h, WihH, WhhH, bsum,
      mlpW1T, mlp_b1, mlp_W2, mlp_b2, scores);

  loss_kernel<<<1, 1024, 0, stream>>>(label, scores, out);
}
